// GPT_15642270892287
// MI455X (gfx1250) — matrix operand order repaired and checked by reading, not compiled, not run on hardware
//
#include <hip/hip_runtime.h>
#include <math.h>

// ---------------- model constants ----------------
constexpr int Bn = 4, Tn = 1024, Vn = 50257, Cn = 768, Hn = 12, Ln = 4;
constexpr int Dn = Cn / Hn;          // 64
constexpr int BT = Bn * Tn;          // 4096

typedef __attribute__((ext_vector_type(16))) __bf16 v16bf;
typedef __attribute__((ext_vector_type(8)))  __bf16 v8bf;
typedef __attribute__((ext_vector_type(8)))  float  v8f;
typedef __attribute__((ext_vector_type(4)))  unsigned v4u;
typedef __attribute__((ext_vector_type(4)))  int      v4i;
typedef __attribute__((ext_vector_type(8)))  int      v8i;

static __device__ inline v16bf cat8(v8bf lo, v8bf hi) {
  return __builtin_shufflevector(lo, hi, 0,1,2,3,4,5,6,7,8,9,10,11,12,13,14,15);
}
static __device__ inline float gelu_f(float x) {
  float x3 = x * x * x;
  return 0.5f * x * (1.f + tanhf(0.79788456f * (x + 0.044715f * x3)));
}

// ---------------- TDM: 2D bf16 tile (128 rows x 32 cols) global -> LDS ----
// LDS padding: 4 DWORDs after every 16 DWORDs -> row stride 40 bf16 (80 B).
static __device__ inline void tdm_load_tileA(const __bf16* gsrc, unsigned lds_off,
                                             int K /* elements per row & stride */) {
  unsigned long long ga = (unsigned long long)(uintptr_t)gsrc;
  v4u g0;
  g0[0] = 1u;                                        // count=1, no gather
  g0[1] = lds_off;                                   // lds_addr (bytes)
  g0[2] = (unsigned)(ga & 0xffffffffu);              // global_addr[31:0]
  g0[3] = (unsigned)((ga >> 32) & 0x01ffffffu) | (2u << 30);  // addr[56:32] | type=2
  v8i g1;
  const unsigned dim1 = 1u << 20;                    // plenty of rows (no OOB)
  g1[0] = (int)((1u << 16)                           // data_size = 2 bytes
              | (1u << 20)                           // pad_enable
              | (3u << 22)                           // pad_interval: 16 DWORDs
              | (3u << 25));                         // pad_amount: 4 DWORDs
  g1[1] = (int)(((unsigned)K & 0xffffu) << 16);      // tensor_dim0[15:0]
  g1[2] = (int)((((unsigned)K >> 16) & 0xffffu) | ((dim1 & 0xffffu) << 16));
  g1[3] = (int)(((dim1 >> 16) & 0xffffu) | (32u << 16));   // tile_dim0 = 32
  g1[4] = (int)(128u);                               // tile_dim1 = 128, tile_dim2 = 0
  g1[5] = (int)K;                                    // tensor_dim0_stride lo
  g1[6] = 0; g1[7] = 0;
  v4i gz = {0, 0, 0, 0};
#if defined(__clang_major__) && __clang_major__ >= 23
  v8i gz8 = {0, 0, 0, 0, 0, 0, 0, 0};
  __builtin_amdgcn_tensor_load_to_lds(g0, g1, gz, gz, gz8, 0);
#else
  __builtin_amdgcn_tensor_load_to_lds(g0, g1, gz, gz, 0);
#endif
}

// ---------------- embedding ----------------
__global__ __launch_bounds__(256) void embed_kernel(
    const int* __restrict__ idx, const float* __restrict__ wte,
    const float* __restrict__ wpe, float* __restrict__ x) {
  int i = blockIdx.x * 256 + threadIdx.x;            // over BT*C
  int bt = i / Cn, c = i - bt * Cn;
  int t = bt & (Tn - 1);
  x[i] = wte[(size_t)idx[bt] * Cn + c] + wpe[(size_t)t * Cn + c];
}

// ---------------- fp32 -> bf16 transpose of a weight matrix ----------------
// W[K][N] (f32) -> Wt[N][K] (bf16).  block (32,8), grid (ceil(N/32), ceil(K/32))
__global__ __launch_bounds__(256) void wtr_kernel(
    const float* __restrict__ W, __bf16* __restrict__ Wt, int K, int N) {
  __shared__ float tile[32][33];
  const int tx = threadIdx.x, ty = threadIdx.y;
  int n = blockIdx.x * 32 + tx;
#pragma unroll
  for (int j = 0; j < 32; j += 8) {
    int k = blockIdx.y * 32 + ty + j;
    if (n < N && k < K) tile[ty + j][tx] = W[(size_t)k * N + n];
  }
  __syncthreads();
  int k2 = blockIdx.y * 32 + tx;
#pragma unroll
  for (int j = 0; j < 32; j += 8) {
    int n2 = blockIdx.x * 32 + ty + j;
    if (k2 < K && n2 < N) Wt[(size_t)n2 * K + k2] = (__bf16)tile[tx][ty + j];
  }
}

// ---------------- V transpose: qkv[b][t][2C + h*D + d] -> vT[b][h][d][t] ----
// Makes attention's WMMA B-fragment (two runs of 8 consecutive keys per lane) two b128
// loads instead of 16 strided u16 gathers.  block (32,8), grid (T/32, 2*B*H)
__global__ __launch_bounds__(256) void vtr_kernel(
    const __bf16* __restrict__ qkv, __bf16* __restrict__ vT) {
  __shared__ __bf16 tile[32][33];
  const int tx = threadIdx.x, ty = threadIdx.y;
  int bh = blockIdx.y >> 1, dt = blockIdx.y & 1;
  int b = bh / Hn, h = bh - b * Hn;
  int t0 = blockIdx.x * 32, d0 = dt * 32;
  const __bf16* src = qkv + (size_t)b * Tn * (3 * Cn) + 2 * Cn + h * Dn;
#pragma unroll
  for (int j = 0; j < 32; j += 8)
    tile[ty + j][tx] = src[(size_t)(t0 + ty + j) * (3 * Cn) + d0 + tx];
  __syncthreads();
  __bf16* dst = vT + ((size_t)(b * Hn + h) * Dn) * Tn;
#pragma unroll
  for (int j = 0; j < 32; j += 8)
    dst[(size_t)(d0 + ty + j) * Tn + t0 + tx] = tile[tx][ty + j];
}

// ---------------- layernorm (one row / block), bf16 out ----------------
__global__ __launch_bounds__(256) void ln_kernel(
    const float* __restrict__ x, const float* __restrict__ g,
    const float* __restrict__ b, __bf16* __restrict__ out) {
  __shared__ float ssum[256], ssq[256];
  const int tid = threadIdx.x;
  const size_t row = blockIdx.x;
  const float* xr = x + row * Cn;
  float s = 0.f, q = 0.f;
  for (int i = tid; i < Cn; i += 256) { float v = xr[i]; s += v; q += v * v; }
  ssum[tid] = s; ssq[tid] = q;
  __syncthreads();
  for (int st = 128; st > 0; st >>= 1) {
    if (tid < st) { ssum[tid] += ssum[tid + st]; ssq[tid] += ssq[tid + st]; }
    __syncthreads();
  }
  float mu = ssum[0] * (1.f / Cn);
  float var = ssq[0] * (1.f / Cn) - mu * mu;
  float rs = rsqrtf(var + 1e-5f);
  for (int i = tid; i < Cn; i += 256)
    out[row * Cn + i] = (__bf16)((xr[i] - mu) * rs * g[i] + b[i]);
}

// ---------------- bf16 WMMA GEMM, TDM-staged A, direct-global B ----------
// C[M,N] = A[M,K](bf16) @ Wt[N,K](bf16, pre-transposed) + bias (+gelu)(+resid)
// block 256 = 8 waves; tile 128x128; BK=32; wave tile 32x64 (2x4 WMMA)
constexpr int LDSPAD = 40;  // bf16 row stride produced by TDM padding

template <bool GELU, bool RESID, bool OUTBF16>
__global__ __launch_bounds__(256) void gemm_kernel(
    const __bf16* __restrict__ A, const __bf16* __restrict__ Wt,
    const float* __restrict__ bias, const float* __restrict__ resid,
    float* __restrict__ Cf, __bf16* __restrict__ Cb,
    int M, int N, int K) {
  __shared__ __align__(16) __bf16 As[2][128 * LDSPAD];

  const int t = threadIdx.x;
  const int wave = t >> 5, lane = t & 31;
  const int hf = lane >> 4, l16 = lane & 15;
  const int wm = wave >> 1, wn = wave & 1;           // 4 x 2 wave grid
  const int m0 = blockIdx.y * 128, n0 = blockIdx.x * 128;

  v8f acc[2][4];
#pragma unroll
  for (int mt = 0; mt < 2; ++mt)
#pragma unroll
    for (int nt = 0; nt < 4; ++nt)
#pragma unroll
      for (int r = 0; r < 8; ++r) acc[mt][nt][r] = 0.f;

  const int niter = K >> 5;
  if (wave == 0)
    tdm_load_tileA(A + (size_t)m0 * K, (unsigned)(uintptr_t)&As[0][0], K);

  for (int i = 0; i < niter; ++i) {
    const int k0 = i << 5;
    if (wave == 0) {
      if (i + 1 < niter) {
        tdm_load_tileA(A + (size_t)m0 * K + (k0 + 32),
                       (unsigned)(uintptr_t)&As[(i + 1) & 1][0], K);
        __builtin_amdgcn_s_wait_tensorcnt(1);   // tile i resident
      } else {
        __builtin_amdgcn_s_wait_tensorcnt(0);
      }
    }
    __syncthreads();

    const __bf16* Abuf = As[i & 1];
    v16bf af[2], bfm[4];
#pragma unroll
    for (int mt = 0; mt < 2; ++mt) {
      int row = wm * 32 + mt * 16 + l16;
      const __bf16* p = &Abuf[row * LDSPAD + 8 * hf];
      af[mt] = cat8(*(const v8bf*)p, *(const v8bf*)(p + 16));
    }
#pragma unroll
    for (int nt = 0; nt < 4; ++nt) {
      int col = n0 + wn * 64 + nt * 16 + l16;
      if (col < N) {
        const __bf16* p = Wt + (size_t)col * K + k0 + 8 * hf;
        bfm[nt] = cat8(*(const v8bf*)p, *(const v8bf*)(p + 16));
        if (k0 + 32 < K) __builtin_prefetch(p + 32, 0, 1);
      } else {
#pragma unroll
        for (int e = 0; e < 16; ++e) bfm[nt][e] = (__bf16)0.f;
      }
    }
#pragma unroll
    for (int mt = 0; mt < 2; ++mt)
#pragma unroll
      for (int nt = 0; nt < 4; ++nt)
        acc[mt][nt] = __builtin_amdgcn_wmma_f32_16x16x32_bf16(
            false, af[mt], false, bfm[nt], (short)0, acc[mt][nt], false, false);
    __syncthreads();
  }

  // ---- epilogue ----
#pragma unroll
  for (int mt = 0; mt < 2; ++mt)
#pragma unroll
    for (int nt = 0; nt < 4; ++nt) {
      int col = n0 + wn * 64 + nt * 16 + l16;
      if (col < N) {
        float bv = bias[col];
#pragma unroll
        for (int r = 0; r < 8; ++r) {
          int row = m0 + wm * 32 + mt * 16 + r + 8 * hf;
          float v = acc[mt][nt][r] + bv;
          if (GELU) v = gelu_f(v);
          if (RESID) v += resid[(size_t)row * N + col];
          if (OUTBF16) Cb[(size_t)row * N + col] = (__bf16)v;
          else         Cf[(size_t)row * N + col] = v;
        }
      }
    }
}

// ---------------- WMMA flash attention (bf16 in, bf16 out) ----------------
// Computes S^T = K·Q^T so softmax'd P lands directly in the bf16 A-matrix
// register layout for O += P·V (zero cross-lane transpose).  V comes from
// the pre-transposed vT[b][h][d][t] so every operand load is a b128.
__global__ __launch_bounds__(128) void attn_kernel(
    const __bf16* __restrict__ qkv, const __bf16* __restrict__ vT,
    __bf16* __restrict__ out) {
  const int wave = threadIdx.x >> 5, lane = threadIdx.x & 31;
  const int hf = lane >> 4, l16 = lane & 15;
  const int QT = Tn / 16;                             // 64 q-tiles per (b,h)
  int tile = blockIdx.x * 4 + wave;                   // 3072 tiles total
  int bh = tile / QT, qt = tile - bh * QT;
  int b = bh / Hn, h = bh - b * Hn;
  int qb = qt * 16;

  const size_t rstride = 3 * Cn;                      // 2304 bf16 per token
  const __bf16* qbase = qkv + (size_t)b * Tn * rstride + (size_t)h * Dn;
  const __bf16* kbase = qbase + Cn;
  const __bf16* vTh = vT + ((size_t)(b * Hn + h) * Dn) * Tn;

  v16bf bq[2];
  {
    const __bf16* qr = qbase + (size_t)(qb + l16) * rstride;
#pragma unroll
    for (int c = 0; c < 2; ++c) {
      const __bf16* p = qr + 32 * c + 8 * hf;
      bq[c] = cat8(*(const v8bf*)p, *(const v8bf*)(p + 16));
    }
  }

  float m = -3.0e38f, lsumr = 0.f;
  v8f o[4];
#pragma unroll
  for (int c = 0; c < 4; ++c)
#pragma unroll
    for (int r = 0; r < 8; ++r) o[c][r] = 0.f;

  for (int kb = 0; kb <= qb; kb += 32) {
    v8f st[2];
#pragma unroll
    for (int g = 0; g < 2; ++g) {
#pragma unroll
      for (int r = 0; r < 8; ++r) st[g][r] = 0.f;
#pragma unroll
      for (int c = 0; c < 2; ++c) {
        const __bf16* kr =
            kbase + (size_t)(kb + g * 16 + l16) * rstride + 32 * c + 8 * hf;
        v16bf ak = cat8(*(const v8bf*)kr, *(const v8bf*)(kr + 16));
        st[g] = __builtin_amdgcn_wmma_f32_16x16x32_bf16(
            false, ak, false, bq[c], (short)0, st[g], false, false);
      }
    }
    const int q = qb + l16;
    float mloc = -3.0e38f;
    if (kb < qb) {  // uniform: no per-element causal mask, no cmps (keys > q unmasked when kb == qb-16)
#pragma unroll
      for (int g = 0; g < 2; ++g)
#pragma unroll
        for (int r = 0; r < 8; ++r) {
          float s = st[g][r] * 0.125f;
          st[g][r] = s;
          mloc = fmaxf(mloc, s);
        }
    } else {        // diagonal block: per-element causal mask
#pragma unroll
      for (int g = 0; g < 2; ++g)
#pragma unroll
        for (int r = 0; r < 8; ++r) {
          int key = kb + g * 16 + r + 8 * hf;
          float s = (key <= q) ? st[g][r] * 0.125f : -3.0e38f;
          st[g][r] = s;
          mloc = fmaxf(mloc, s);
        }
    }
    mloc = fmaxf(mloc, __shfl_xor(mloc, 16, 32));
    float mnew = fmaxf(m, mloc);
    float alpha = __expf(m - mnew);
    float ps = 0.f;
    v16bf ap;
#pragma unroll
    for (int g = 0; g < 2; ++g)
#pragma unroll
      for (int r = 0; r < 8; ++r) {
        float p = __expf(st[g][r] - mnew);
        ps += p;
        ap[g * 8 + r] = (__bf16)p;
      }
    ps += __shfl_xor(ps, 16, 32);
    lsumr = lsumr * alpha + ps;
    m = mnew;
    float al[8];
#pragma unroll
    for (int r = 0; r < 8; ++r) al[r] = __shfl(alpha, r + 8 * hf, 32);
#pragma unroll
    for (int c = 0; c < 4; ++c)
#pragma unroll
      for (int r = 0; r < 8; ++r) o[c][r] *= al[r];
    // O += P · V :  lane d-column = 16c+l16, keys kb+8hf..+7 and kb+16+8hf..+7 in vT
#pragma unroll
    for (int c = 0; c < 4; ++c) {
      const __bf16* p = vTh + (size_t)(16 * c + l16) * Tn + kb + 8 * hf;
      v16bf bv = cat8(*(const v8bf*)p, *(const v8bf*)(p + 16));
      o[c] = __builtin_amdgcn_wmma_f32_16x16x32_bf16(
          false, ap, false, bv, (short)0, o[c], false, false);
    }
  }

  float li[8];
#pragma unroll
  for (int r = 0; r < 8; ++r) li[r] = 1.f / __shfl(lsumr, r + 8 * hf, 32);
#pragma unroll
  for (int c = 0; c < 4; ++c)
#pragma unroll
    for (int r = 0; r < 8; ++r) {
      int row = qb + r + 8 * hf;
      out[((size_t)(b * Tn + row)) * Cn + h * Dn + 16 * c + l16] =
          (__bf16)(o[c][r] * li[r]);
    }
}

// ---------------- host driver ----------------
extern "C" void kernel_launch(void* const* d_in, const int* in_sizes, int n_in,
                              void* d_out, int out_size, void* d_ws,
                              size_t ws_size, hipStream_t stream) {
  const int*   idx     = (const int*)d_in[0];
  const float* wte     = (const float*)d_in[1];
  const float* wpe     = (const float*)d_in[2];
  const float* ln1_g   = (const float*)d_in[3];
  const float* ln1_b   = (const float*)d_in[4];
  const float* attn_w  = (const float*)d_in[5];
  const float* attn_b  = (const float*)d_in[6];
  const float* aproj_w = (const float*)d_in[7];
  const float* aproj_b = (const float*)d_in[8];
  const float* ln2_g   = (const float*)d_in[9];
  const float* ln2_b   = (const float*)d_in[10];
  const float* fc_w    = (const float*)d_in[11];
  const float* fc_b    = (const float*)d_in[12];
  const float* proj_w  = (const float*)d_in[13];
  const float* proj_b  = (const float*)d_in[14];
  const float* lnf_g   = (const float*)d_in[15];
  const float* lnf_b   = (const float*)d_in[16];
  const float* head_w  = (const float*)d_in[17];
  const float* head_b  = (const float*)d_in[18];
  float* logits = (float*)d_out;

  // workspace carve-up (~203 MB)
  char* ws = (char*)d_ws;
  size_t off = 0;
  auto carve = [&](size_t bytes) {
    char* p = ws + off;
    off = (off + bytes + 255) & ~(size_t)255;
    return p;
  };
  float*  x     = (float*) carve((size_t)BT * Cn * 4);        // residual (f32)
  __bf16* lnb   = (__bf16*)carve((size_t)BT * Cn * 2);        // LN / attn out
  __bf16* qkvb  = (__bf16*)carve((size_t)BT * 3 * Cn * 2);    // qkv bf16
  __bf16* vT    = (__bf16*)carve((size_t)BT * Cn * 2);        // V transposed
  __bf16* h2    = (__bf16*)carve((size_t)BT * 4 * Cn * 2);    // MLP hidden bf16
  __bf16* watt  = (__bf16*)carve((size_t)Ln * 3 * Cn * Cn * 2);
  __bf16* wapr  = (__bf16*)carve((size_t)Ln * Cn * Cn * 2);
  __bf16* wfc   = (__bf16*)carve((size_t)Ln * 4 * Cn * Cn * 2);
  __bf16* wproj = (__bf16*)carve((size_t)Ln * Cn * 4 * Cn * 2);
  __bf16* whead = (__bf16*)carve((size_t)Vn * Cn * 2);
  (void)ws_size; (void)in_sizes; (void)n_in; (void)out_size;

  dim3 blk(256);
  dim3 tblk(32, 8);
  // ---- pre-transpose + bf16-convert all weights: W[K][N] -> Wt[N][K] ----
  for (int l = 0; l < Ln; ++l) {
    wtr_kernel<<<dim3((3 * Cn) / 32, Cn / 32), tblk, 0, stream>>>(
        attn_w + (size_t)l * Cn * 3 * Cn, watt + (size_t)l * 3 * Cn * Cn, Cn, 3 * Cn);
    wtr_kernel<<<dim3(Cn / 32, Cn / 32), tblk, 0, stream>>>(
        aproj_w + (size_t)l * Cn * Cn, wapr + (size_t)l * Cn * Cn, Cn, Cn);
    wtr_kernel<<<dim3((4 * Cn) / 32, Cn / 32), tblk, 0, stream>>>(
        fc_w + (size_t)l * Cn * 4 * Cn, wfc + (size_t)l * 4 * Cn * Cn, Cn, 4 * Cn);
    wtr_kernel<<<dim3(Cn / 32, (4 * Cn) / 32), tblk, 0, stream>>>(
        proj_w + (size_t)l * 4 * Cn * Cn, wproj + (size_t)l * Cn * 4 * Cn, 4 * Cn, Cn);
  }
  wtr_kernel<<<dim3((Vn + 31) / 32, Cn / 32), tblk, 0, stream>>>(
      head_w, whead, Cn, Vn);

  embed_kernel<<<(BT * Cn) / 256, blk, 0, stream>>>(idx, wte, wpe, x);

  for (int l = 0; l < Ln; ++l) {
    ln_kernel<<<BT, blk, 0, stream>>>(x, ln1_g + l * Cn, ln1_b + l * Cn, lnb);
    gemm_kernel<false, false, true><<<dim3((3 * Cn) / 128, BT / 128), blk, 0, stream>>>(
        lnb, watt + (size_t)l * 3 * Cn * Cn, attn_b + (size_t)l * 3 * Cn,
        nullptr, nullptr, qkvb, BT, 3 * Cn, Cn);
    vtr_kernel<<<dim3(Tn / 32, 2 * Bn * Hn), tblk, 0, stream>>>(qkvb, vT);
    attn_kernel<<<(Bn * Hn * (Tn / 16)) / 4, dim3(128), 0, stream>>>(qkvb, vT, lnb);
    gemm_kernel<false, true, false><<<dim3(Cn / 128, BT / 128), blk, 0, stream>>>(
        lnb, wapr + (size_t)l * Cn * Cn, aproj_b + (size_t)l * Cn,
        x, x, nullptr, BT, Cn, Cn);
    ln_kernel<<<BT, blk, 0, stream>>>(x, ln2_g + l * Cn, ln2_b + l * Cn, lnb);
    gemm_kernel<true, false, true><<<dim3((4 * Cn) / 128, BT / 128), blk, 0, stream>>>(
        lnb, wfc + (size_t)l * 4 * Cn * Cn, fc_b + (size_t)l * 4 * Cn,
        nullptr, nullptr, h2, BT, 4 * Cn, Cn);
    gemm_kernel<false, true, false><<<dim3(Cn / 128, BT / 128), blk, 0, stream>>>(
        h2, wproj + (size_t)l * Cn * 4 * Cn, proj_b + (size_t)l * Cn,
        x, x, nullptr, BT, Cn, 4 * Cn);
  }
  ln_kernel<<<BT, blk, 0, stream>>>(x, lnf_g, lnf_b, lnb);
  gemm_kernel<false, false, false><<<dim3((Vn + 127) / 128, BT / 128), blk, 0, stream>>>(
      lnb, whead, head_b, nullptr, logits, nullptr, BT, Vn, Cn);
}
